// CMEncoder_23502061043973
// MI455X (gfx1250) — matrix operand order repaired and checked by reading, not compiled, not run on hardware
//
#include <hip/hip_runtime.h>
#include <hip/hip_bf16.h>

typedef _Float16 half_t;
typedef __attribute__((ext_vector_type(16))) _Float16 v16h;
typedef __attribute__((ext_vector_type(8)))  _Float16 v8h;
typedef __attribute__((ext_vector_type(8)))  float    v8f;

union V16 { v16h v; v8h h[2]; };

// ---------------------------------------------------------------------------
// WMMA helpers (CDNA5 gfx1250, wave32)
// ---------------------------------------------------------------------------
__device__ __forceinline__ v8f wmma_f16f32(v16h a, v16h b, v8f c) {
  // D = A(16x32,f16) * B(32x16,f16) + C(16x16,f32)
  return __builtin_amdgcn_wmma_f32_16x16x32_f16(false, a, false, b, (short)0, c,
                                                false, false);
}

// A operand: 16(M) x 32(K) tile from row-major [M][ldk] f16 source.
// HW layout: lanes 0-15 row m, K slots {0..7,16..23}; lanes 16-31 row m, K {8..15,24..31}.
__device__ __forceinline__ v16h load_a_tile(const half_t* base, int ldk, int m0,
                                            int k0, int lane) {
  const half_t* p = base + (size_t)(m0 + (lane & 15)) * ldk + k0 + ((lane >> 4) << 3);
  V16 r;
  r.h[0] = *(const v8h*)(p);
  r.h[1] = *(const v8h*)(p + 16);
  return r.v;
}

// B operand: 32(K) x 16(N) tile; source stored row-major [N][ldk] (i.e. B^T).
// HW layout: lanes 0-15 col n, K slots {0..7,16..23}; lanes 16-31 col n, K {8..15,24..31}.
__device__ __forceinline__ v16h load_b_tile(const half_t* base, int ldk, int n0,
                                            int k0, int lane) {
  const half_t* p = base + (size_t)(n0 + (lane & 15)) * ldk + k0 + ((lane >> 4) << 3);
  V16 r;
  r.h[0] = *(const v8h*)(p);
  r.h[1] = *(const v8h*)(p + 16);
  return r.v;
}

__device__ __forceinline__ float gelu_exact(float v) {
  return 0.5f * v * (1.0f + erff(v * 0.70710678118654752f));
}

// ---------------------------------------------------------------------------
// fp32 [R][Cc] -> f16 transpose [Cc][R]
// ---------------------------------------------------------------------------
__global__ __launch_bounds__(256) void transpose_cvt_kernel(
    const float* __restrict__ in, half_t* __restrict__ outT, int R, int Cc) {
  __shared__ float tile[32][33];
  int cb = blockIdx.x * 32, rb = blockIdx.y * 32;
  int tx = threadIdx.x, ty = threadIdx.y;  // (32,8)
  #pragma unroll
  for (int i = 0; i < 32; i += 8)
    tile[ty + i][tx] = in[(size_t)(rb + ty + i) * Cc + cb + tx];
  __syncthreads();
  #pragma unroll
  for (int i = 0; i < 32; i += 8)
    outT[(size_t)(cb + ty + i) * R + rb + tx] = (half_t)tile[tx][ty + i];
}

__global__ __launch_bounds__(256) void cvt_f16_kernel(const float* __restrict__ in,
                                                      half_t* __restrict__ out, int n) {
  int i = blockIdx.x * blockDim.x + threadIdx.x;
  if (i < n) out[i] = (half_t)in[i];
}

// ---------------------------------------------------------------------------
// Generic NT GEMM: D[n][m] = sum_k A[m][k]*B[n][k] (+bias, +act), f16 out.
// Out layout: [N][M] row-major (leading dim M).
// BIAS_MODE 0: bias indexed by m; 1: bias indexed by n. ACT 1: exact GELU.
// ---------------------------------------------------------------------------
template <int BIAS_MODE, int ACT>
__global__ __launch_bounds__(128) void gemm_nt_proj(
    const half_t* __restrict__ A, const half_t* __restrict__ B,
    const float* __restrict__ bias, half_t* __restrict__ Out,
    int M, int N, int K) {
  int lane = threadIdx.x & 31;
  int wid  = threadIdx.x >> 5;
  int tile = blockIdx.x * (blockDim.x >> 5) + wid;
  int ntm = M >> 4;
  int m0 = (tile % ntm) << 4;
  int n0 = (tile / ntm) << 4;
  if (n0 >= N) return;

  v8f acc = {};
  for (int k0 = 0; k0 < K; k0 += 32) {
    v16h a = load_a_tile(A, K, m0, k0, lane);
    v16h b = load_b_tile(B, K, n0, k0, lane);
    acc = wmma_f16f32(a, b, acc);
  }

  int mrow = m0 + ((lane >> 4) << 3);
  half_t* orow = Out + (size_t)(n0 + (lane & 15)) * M + mrow;
  v8h pk;
  #pragma unroll
  for (int r = 0; r < 8; r++) {
    float bvv = (BIAS_MODE == 0) ? bias[mrow + r] : bias[n0 + (lane & 15)];
    float v = acc[r] + bvv;
    if (ACT) v = gelu_exact(v);
    pk[r] = (half_t)v;
  }
  *(v8h*)orow = pk;
}

// ---------------------------------------------------------------------------
// Flash attention: one wave per 16-query tile, online softmax over N=4096 keys.
// Q,K: [tok][256] f16. Vt: [256][4096] f16 (channel-major). att: [tok][256] f16.
// ---------------------------------------------------------------------------
__global__ __launch_bounds__(32) void attn_kernel(
    const half_t* __restrict__ Q, const half_t* __restrict__ Kh,
    const half_t* __restrict__ Vt, half_t* __restrict__ att) {
  __shared__ __align__(16) float  sS[16 * 32];
  __shared__ __align__(16) half_t sP[16 * 32];
  __shared__ __align__(16) float  sScale[16];

  int lane = threadIdx.x & 31;
  int q0 = blockIdx.x << 4;

  v16h qa[8];
  #pragma unroll
  for (int kc = 0; kc < 8; kc++) qa[kc] = load_a_tile(Q, 256, q0, kc * 32, lane);

  v8f acc[16];
  #pragma unroll
  for (int cb = 0; cb < 16; cb++) acc[cb] = {};

  float mrow = -1e30f, lrow = 0.0f;
  const int mbase = (lane >> 4) << 3;
  const int ncol  = lane & 15;

  for (int kb = 0; kb < 4096; kb += 32) {
    if (kb + 32 < 4096)  // hint next K block toward L2/L0 (global_prefetch_b8)
      __builtin_prefetch(Kh + (size_t)(kb + 32 + ncol) * 256 + ((lane >> 4) << 4), 0, 1);

    // scores S[16 q][32 k] = Q * K^T over C=256
    v8f s0 = {}, s1 = {};
    #pragma unroll
    for (int kc = 0; kc < 8; kc++) {
      v16h b0 = load_b_tile(Kh, 256, kb,      kc * 32, lane);
      v16h b1 = load_b_tile(Kh, 256, kb + 16, kc * 32, lane);
      s0 = wmma_f16f32(qa[kc], b0, s0);
      s1 = wmma_f16f32(qa[kc], b1, s1);
    }
    const float scl = 0.0625f;  // 1/sqrt(256)
    #pragma unroll
    for (int r = 0; r < 8; r++) {
      sS[(mbase + r) * 32 + ncol]      = s0[r] * scl;
      sS[(mbase + r) * 32 + ncol + 16] = s1[r] * scl;
    }
    __syncthreads();

    // online softmax: lane l owns query row l
    if (lane < 16) {
      float mx = mrow;
      #pragma unroll
      for (int j = 0; j < 32; j++) mx = fmaxf(mx, sS[lane * 32 + j]);
      float lsum = 0.0f;
      #pragma unroll
      for (int j = 0; j < 32; j++) {
        float p = __expf(sS[lane * 32 + j] - mx);
        lsum += p;
        sP[lane * 32 + j] = (half_t)p;
      }
      float corr = __expf(mrow - mx);
      lrow = lrow * corr + lsum;
      mrow = mx;
      sScale[lane] = corr;
    }
    __syncthreads();

    float cscale[8];
    #pragma unroll
    for (int r = 0; r < 8; r++) cscale[r] = sScale[mbase + r];

    // P (16x32) as A operand, straight from LDS in HW layout
    v16h pa = load_a_tile(sP, 32, 0, 0, lane);

    #pragma unroll
    for (int cb = 0; cb < 16; cb++) {
      #pragma unroll
      for (int r = 0; r < 8; r++) acc[cb][r] *= cscale[r];
      v16h vb = load_b_tile(Vt, 4096, cb * 16, kb, lane);
      acc[cb] = wmma_f16f32(pa, vb, acc[cb]);
    }
    __syncthreads();
  }

  if (lane < 16) sScale[lane] = 1.0f / lrow;
  __syncthreads();
  float fin[8];
  #pragma unroll
  for (int r = 0; r < 8; r++) fin[r] = sScale[mbase + r];

  #pragma unroll
  for (int cb = 0; cb < 16; cb++)
    #pragma unroll
    for (int r = 0; r < 8; r++)
      att[(size_t)(q0 + mbase + r) * 256 + cb * 16 + ncol] = (half_t)(acc[cb][r] * fin[r]);
}

// ---------------------------------------------------------------------------
// z = att*Wo^T + bo, then LayerNorm over channels -> zn [tok][256] f16
// One wave per 16-token row block.
// ---------------------------------------------------------------------------
__global__ __launch_bounds__(32) void zo_ln_kernel(
    const half_t* __restrict__ att, const half_t* __restrict__ Woh,
    const float* __restrict__ bo, const float* __restrict__ lnw,
    const float* __restrict__ lnb, half_t* __restrict__ zn) {
  __shared__ __align__(16) float sZ[16 * 256];
  __shared__ float sMu[16], sRs[16];

  int lane = threadIdx.x & 31;
  int t0 = blockIdx.x << 4;
  const int mbase = (lane >> 4) << 3;
  const int ncol  = lane & 15;

  v16h aa[8];
  #pragma unroll
  for (int kc = 0; kc < 8; kc++) aa[kc] = load_a_tile(att, 256, t0, kc * 32, lane);

  for (int ob = 0; ob < 16; ob++) {
    v8f acc = {};
    #pragma unroll
    for (int kc = 0; kc < 8; kc++)
      acc = wmma_f16f32(aa[kc], load_b_tile(Woh, 256, ob * 16, kc * 32, lane), acc);
    float bvv = bo[ob * 16 + ncol];
    #pragma unroll
    for (int r = 0; r < 8; r++)
      sZ[(mbase + r) * 256 + ob * 16 + ncol] = acc[r] + bvv;
  }
  __syncthreads();

  if (lane < 16) {
    float s = 0.0f, s2 = 0.0f;
    for (int c = 0; c < 256; c++) {
      float v = sZ[lane * 256 + c];
      s += v; s2 += v * v;
    }
    float mu = s * (1.0f / 256.0f);
    float var = s2 * (1.0f / 256.0f) - mu * mu;
    sMu[lane] = mu;
    sRs[lane] = rsqrtf(var + 1e-6f);
  }
  __syncthreads();

  for (int e = lane; e < 16 * 256; e += 32) {
    int t = e >> 8, c = e & 255;
    float v = (sZ[e] - sMu[t]) * sRs[t] * lnw[c] + lnb[c];
    zn[(size_t)(t0 + t) * 256 + c] = (half_t)v;
  }
}

// ---------------------------------------------------------------------------
// out[o][hw] = x[o][hw] + b2[o] + sum_k H[tok][k]*W2[o][k]  (f32 output)
// M = tok, N = o; D lane n = channel, m = token -> contiguous f32 stores.
// ---------------------------------------------------------------------------
__global__ __launch_bounds__(128) void mlp2_kernel(
    const half_t* __restrict__ Hh, const half_t* __restrict__ W2h,
    const float* __restrict__ b2, const float* __restrict__ x,
    float* __restrict__ out) {
  int lane = threadIdx.x & 31;
  int wid  = threadIdx.x >> 5;
  int tile = blockIdx.x * (blockDim.x >> 5) + wid;
  int t0 = (tile & 255) << 4;   // 256 token tiles
  int o0 = (tile >> 8) << 4;    // 16 channel tiles
  if (o0 >= 256) return;

  v8f acc = {};
  for (int k0 = 0; k0 < 512; k0 += 32) {
    v16h a = load_a_tile(Hh, 512, t0, k0, lane);
    v16h b = load_b_tile(W2h, 512, o0, k0, lane);
    acc = wmma_f16f32(a, b, acc);
  }

  int o  = o0 + (lane & 15);
  int tb = t0 + ((lane >> 4) << 3);
  const float* xr = x + (size_t)o * 4096 + tb;
  float* orow = out + (size_t)o * 4096 + tb;
  float bvv = b2[o];
  float4 xa = *(const float4*)xr;
  float4 xb = *(const float4*)(xr + 4);
  float4 r0, r1;
  r0.x = acc[0] + bvv + xa.x; r0.y = acc[1] + bvv + xa.y;
  r0.z = acc[2] + bvv + xa.z; r0.w = acc[3] + bvv + xa.w;
  r1.x = acc[4] + bvv + xb.x; r1.y = acc[5] + bvv + xb.y;
  r1.z = acc[6] + bvv + xb.z; r1.w = acc[7] + bvv + xb.w;
  *(float4*)orow = r0;
  *(float4*)(orow + 4) = r1;
}

// ---------------------------------------------------------------------------
extern "C" void kernel_launch(void* const* d_in, const int* in_sizes, int n_in,
                              void* d_out, int out_size, void* d_ws, size_t ws_size,
                              hipStream_t stream) {
  (void)in_sizes; (void)n_in; (void)out_size; (void)ws_size;
  const float* x   = (const float*)d_in[0];
  const float* y   = (const float*)d_in[1];
  const float* Wq  = (const float*)d_in[2];
  const float* bq  = (const float*)d_in[3];
  const float* Wk  = (const float*)d_in[4];
  const float* bk  = (const float*)d_in[5];
  const float* Wv  = (const float*)d_in[6];
  const float* bv  = (const float*)d_in[7];
  const float* Wo  = (const float*)d_in[8];
  const float* bo  = (const float*)d_in[9];
  const float* lnw = (const float*)d_in[10];
  const float* lnb = (const float*)d_in[11];
  const float* W1  = (const float*)d_in[12];
  const float* b1  = (const float*)d_in[13];
  const float* W2  = (const float*)d_in[14];
  const float* b2  = (const float*)d_in[15];
  float* out = (float*)d_out;

  char* ws = (char*)d_ws;
  const size_t MB = 1024 * 1024;
  half_t* xt  = (half_t*)(ws + 0 * MB);            // [4096][256]
  half_t* yt  = (half_t*)(ws + 2 * MB);            // [4096][256]
  half_t* Qh  = (half_t*)(ws + 4 * MB);            // [4096][256]
  half_t* Kh  = (half_t*)(ws + 6 * MB);            // [4096][256]
  half_t* Vt  = (half_t*)(ws + 8 * MB);            // [256][4096]
  half_t* att = (half_t*)(ws + 10 * MB);           // [4096][256]
  half_t* zn  = (half_t*)(ws + 12 * MB);           // [4096][256]
  half_t* Hh  = (half_t*)(ws + 14 * MB);           // [4096][512]
  half_t* Wqh = (half_t*)(ws + 18 * MB);
  half_t* Wkh = (half_t*)(ws + 18 * MB + 128 * 1024);
  half_t* Wvh = (half_t*)(ws + 18 * MB + 256 * 1024);
  half_t* Woh = (half_t*)(ws + 18 * MB + 384 * 1024);
  half_t* W1h = (half_t*)(ws + 18 * MB + 512 * 1024);
  half_t* W2h = (half_t*)(ws + 18 * MB + 768 * 1024);

  // Stage 0: transposes + f16 conversion
  transpose_cvt_kernel<<<dim3(128, 8), dim3(32, 8), 0, stream>>>(x, xt, 256, 4096);
  transpose_cvt_kernel<<<dim3(128, 8), dim3(32, 8), 0, stream>>>(y, yt, 256, 4096);
  cvt_f16_kernel<<<256, 256, 0, stream>>>(Wq, Wqh, 65536);
  cvt_f16_kernel<<<256, 256, 0, stream>>>(Wk, Wkh, 65536);
  cvt_f16_kernel<<<256, 256, 0, stream>>>(Wv, Wvh, 65536);
  cvt_f16_kernel<<<256, 256, 0, stream>>>(Wo, Woh, 65536);
  cvt_f16_kernel<<<512, 256, 0, stream>>>(W1, W1h, 131072);
  cvt_f16_kernel<<<512, 256, 0, stream>>>(W2, W2h, 131072);

  // Stage 1: Q/K/V projections (WMMA)
  gemm_nt_proj<0, 0><<<1024, 128, 0, stream>>>(Wqh, xt, bq, Qh, 256, 4096, 256);
  gemm_nt_proj<0, 0><<<1024, 128, 0, stream>>>(Wkh, yt, bk, Kh, 256, 4096, 256);
  gemm_nt_proj<1, 0><<<1024, 128, 0, stream>>>(yt, Wvh, bv, Vt, 4096, 256, 256);

  // Stage 2: flash attention (WMMA + online softmax)
  attn_kernel<<<256, 32, 0, stream>>>(Qh, Kh, Vt, att);

  // Stage 3: output projection + LayerNorm2d
  zo_ln_kernel<<<256, 32, 0, stream>>>(att, Woh, bo, lnw, lnb, zn);

  // Stage 4: MLP
  gemm_nt_proj<0, 1><<<2048, 128, 0, stream>>>(W1h, zn, b1, Hh, 512, 4096, 256);
  mlp2_kernel<<<1024, 128, 0, stream>>>(Hh, W2h, b2, x, out);
}
